// AlternatingSimple_90340342104454
// MI455X (gfx1250) — matrix operand order repaired and checked by reading, not compiled, not run on hardware
//
#include <hip/hip_runtime.h>
#include <hip/hip_bf16.h>
#include <stdint.h>

#define DEV __device__ __forceinline__

static constexpr int NN = 50000;   // nodes
static constexpr int NE = 800000;  // edges
static constexpr int NB = 128;     // batch / globals

typedef __attribute__((ext_vector_type(16))) __bf16       v16bf;
typedef __attribute__((ext_vector_type(8)))  float        v8f;
typedef __attribute__((ext_vector_type(4)))  float        f32x4;
typedef __attribute__((ext_vector_type(4)))  unsigned int u32x4;

union FragAB { v16bf v; u32x4 q[2]; unsigned short s[16]; };

DEV v8f wmma_bf16(v16bf a, v16bf b, v8f c) {
    return __builtin_amdgcn_wmma_f32_16x16x32_bf16(false, a, false, b, (short)0, c, false, false);
}

DEV unsigned short f2bf(float f) {
    union { float f; unsigned u; } x; x.f = f;
    unsigned u = x.u + 0x7fffu + ((x.u >> 16) & 1u);   // round-nearest-even
    return (unsigned short)(u >> 16);
}

// packed bf16 subtract: d = a - b (2 elements per dword) via VOP3P v_pk_add_bf16
DEV unsigned int pk_sub_bf16(unsigned int a, unsigned int b) {
    unsigned int d;
    asm volatile("v_pk_add_bf16 %0, %1, %2 neg_lo:[0,1] neg_hi:[0,1]"
                 : "=v"(d) : "v"(a), "v"(b));
    return d;
}

// A fragment (16x32, MxK) from an LDS tile: lane&15 = row, lane>>4 picks K-half.
// slots 0..7 -> K = kBase + half*8 .. +7 ; slots 8..15 -> +16.
DEV v16bf load_a(const unsigned short* sA, int stride, int rowBase, int kBase, int lane) {
    const unsigned short* r = sA + (rowBase + (lane & 15)) * stride + kBase + ((lane >> 4) << 3);
    FragAB f;
    f.q[0] = *(const u32x4*)(r);
    f.q[1] = *(const u32x4*)(r + 16);
    return f.v;
}

// B fragment (32x16, KxN) from packed weights: 512 shorts/fragment, 16/lane contiguous.
DEV v16bf load_b(const unsigned short* Wp, int frag, int lane) {
    const u32x4* p = (const u32x4*)(Wp + (size_t)frag * 512 + lane * 16);
    FragAB f; f.q[0] = p[0]; f.q[1] = p[1];
    return f.v;
}

// ---------------- weight packing: W (K x N, f32 row-major) -> per-fragment bf16 ----
__global__ void pack_w_kernel(const float* __restrict__ W, unsigned short* __restrict__ Wp,
                              int K, int N) {
    int frag = blockIdx.x;         // frag = k_tile * n_tiles + n_tile
    int lane = threadIdx.x;        // 0..31
    int n_tiles = N >> 4;
    int kt = frag / n_tiles, nt = frag % n_tiles;
    int col = nt * 16 + (lane & 15);
    int kb  = kt * 32 + ((lane >> 4) << 3);      // slots 0..7: K half*8..+7, slots 8..15: +16
    unsigned short* o = Wp + (size_t)frag * 512 + lane * 16;
#pragma unroll
    for (int s = 0; s < 16; ++s)
        o[s] = f2bf(W[(size_t)(kb + (s < 8 ? s : s + 8)) * N + col]);
}

// ---------------- misc small kernels ------------------------------------------------
__global__ void zero_f32_kernel(float* p, int n) {
    int i = blockIdx.x * 256 + threadIdx.x;
    if (i < n) p[i] = 0.0f;
}
__global__ void copy_f32_kernel(const float* s, float* d, int n) {
    int i = blockIdx.x * 256 + threadIdx.x;
    if (i < n) d[i] = s[i];
}
__global__ void cvt_f32_bf16_kernel(const float* s, unsigned short* d, int n) {
    int i = blockIdx.x * 256 + threadIdx.x;
    if (i < n) d[i] = f2bf(s[i]);
}
// xc[n] = bf16[ x[n] (64) | u_other[batch[n]] (64) ]
__global__ void build_xc_kernel(const float* __restrict__ x_f32,
                                const unsigned short* __restrict__ x_bf, int x_is_bf,
                                const float* __restrict__ u_other,
                                const int* __restrict__ batch,
                                unsigned short* __restrict__ xc, int n_nodes) {
    int i = blockIdx.x * 256 + threadIdx.x;
    if (i >= n_nodes * 128) return;
    int n = i >> 7, j = i & 127;
    unsigned short v;
    if (j < 64)
        v = x_is_bf ? x_bf[(size_t)n * 64 + j] : f2bf(x_f32[(size_t)n * 64 + j]);
    else
        v = f2bf(u_other[(size_t)batch[n] * 64 + (j - 64)]);
    xc[i] = v;
}

// ---------------- edge model: e_new = MLP([xc[dst]-xc[src], e, u[batch[src]]]) ------
__global__ __launch_bounds__(128) void edge_kernel(
    const unsigned short* __restrict__ xc,
    const float* __restrict__ e_f32, const unsigned short* __restrict__ e_bf, int e_is_bf,
    const unsigned short* __restrict__ u_bf,
    const int* __restrict__ ei, const int* __restrict__ batch,
    const unsigned short* __restrict__ W1p, const float* __restrict__ b1,
    const unsigned short* __restrict__ W2p, const float* __restrict__ b2,
    unsigned short* __restrict__ e_out, float* __restrict__ agg, float* __restrict__ cnt,
    int n_edges) {

    __shared__ unsigned short sA[64 * 264];   // 64 edges x 256 feats (bf16), padded stride
    __shared__ unsigned short sH[64 * 136];   // hidden 128, padded stride
    __shared__ int sDst[64];

    const int tid = threadIdx.x;
    const int lane = tid & 31, wave = tid >> 5;
    const int base = blockIdx.x * 64;

    // ---- phase 1: gather features (wave-local rows) ----
    {
        int el = tid >> 1, half = tid & 1;
        int edge = base + el;
        unsigned short* row = sA + el * 264;
        if (edge < n_edges) {
            int src = ei[edge], dst = ei[n_edges + edge];
            if (half == 0) {
                const unsigned short* pd = xc + (size_t)dst * 128;
                const unsigned short* ps = xc + (size_t)src * 128;
#pragma unroll
                for (int j = 0; j < 128; j += 8) {
                    u32x4 a = *(const u32x4*)(pd + j);
                    u32x4 b = *(const u32x4*)(ps + j);
                    u32x4 d;
                    d.x = pk_sub_bf16(a.x, b.x);
                    d.y = pk_sub_bf16(a.y, b.y);
                    d.z = pk_sub_bf16(a.z, b.z);
                    d.w = pk_sub_bf16(a.w, b.w);
                    *(u32x4*)(row + j) = d;
                }
            } else {
                if (e_is_bf) {
                    const unsigned short* pe = e_bf + (size_t)edge * 64;
#pragma unroll
                    for (int j = 0; j < 64; j += 8)
                        *(u32x4*)(row + 128 + j) = *(const u32x4*)(pe + j);
                } else {
                    const float* pe = e_f32 + (size_t)edge * 64;
#pragma unroll
                    for (int j = 0; j < 64; j += 4) {
                        f32x4 a = *(const f32x4*)(pe + j);
                        row[128 + j + 0] = f2bf(a.x); row[128 + j + 1] = f2bf(a.y);
                        row[128 + j + 2] = f2bf(a.z); row[128 + j + 3] = f2bf(a.w);
                    }
                }
                int bs = batch[src];
                const unsigned short* pu = u_bf + (size_t)bs * 64;
#pragma unroll
                for (int j = 0; j < 64; j += 8)
                    *(u32x4*)(row + 192 + j) = *(const u32x4*)(pu + j);
                sDst[el] = dst;
                atomicAdd(cnt + dst, 1.0f);
            }
        } else {
            u32x4 z = {0u, 0u, 0u, 0u};
            for (int j = half * 128; j < half * 128 + 128; j += 8) *(u32x4*)(row + j) = z;
            if (half) sDst[el] = -1;
        }
    }
    __syncthreads();

    // ---- phase 2: GEMM1 256 -> 128, ReLU ----
    const int rowBase = wave * 16;
#pragma unroll
    for (int nt = 0; nt < 8; ++nt) {
        float bb = b1[nt * 16 + (lane & 15)];
        v8f acc = {bb, bb, bb, bb, bb, bb, bb, bb};
#pragma unroll
        for (int kt = 0; kt < 8; ++kt) {
            v16bf a = load_a(sA, 264, rowBase, kt * 32, lane);
            v16bf b = load_b(W1p, kt * 8 + nt, lane);
            acc = wmma_bf16(a, b, acc);
        }
        int col = nt * 16 + (lane & 15);
        int r0 = rowBase + ((lane >> 4) << 3);
#pragma unroll
        for (int v = 0; v < 8; ++v) {
            float x = acc[v] > 0.f ? acc[v] : 0.f;
            sH[(r0 + v) * 136 + col] = f2bf(x);
        }
    }
    __syncthreads();

    // ---- phase 3: GEMM2 128 -> 64, store e_new + scatter into agg ----
#pragma unroll
    for (int nt = 0; nt < 4; ++nt) {
        float bb = b2[nt * 16 + (lane & 15)];
        v8f acc = {bb, bb, bb, bb, bb, bb, bb, bb};
#pragma unroll
        for (int kt = 0; kt < 4; ++kt) {
            v16bf a = load_a(sH, 136, rowBase, kt * 32, lane);
            v16bf b = load_b(W2p, kt * 4 + nt, lane);
            acc = wmma_bf16(a, b, acc);
        }
        int col = nt * 16 + (lane & 15);
        int r0 = rowBase + ((lane >> 4) << 3);
#pragma unroll
        for (int v = 0; v < 8; ++v) {
            int m = r0 + v;
            int edge = base + m;
            if (edge < n_edges) {
                float val = acc[v];
                e_out[(size_t)edge * 64 + col] = f2bf(val);
                atomicAdd(agg + (size_t)sDst[m] * 64 + col, val);
            }
        }
    }
}

// ---------------- node model + fused attention --------------------------------------
__global__ __launch_bounds__(128) void node_kernel(
    const unsigned short* __restrict__ xc,
    const float* __restrict__ agg, const float* __restrict__ cnt,
    const unsigned short* __restrict__ u_bf, const int* __restrict__ batch,
    const unsigned short* __restrict__ Wn1p, const float* __restrict__ bn1,
    const unsigned short* __restrict__ Wn2p, const float* __restrict__ bn2,
    const unsigned short* __restrict__ Wa1p, const float* __restrict__ ba1,
    const unsigned short* __restrict__ Wa2p, const float* __restrict__ ba2,
    unsigned short* __restrict__ x_new, float* __restrict__ xg, int n_nodes) {

    __shared__ unsigned short sA[64 * 264];
    __shared__ unsigned short sH[64 * 136];
    __shared__ int sB[64];

    const int tid = threadIdx.x;
    const int lane = tid & 31, wave = tid >> 5;
    const int base = blockIdx.x * 64;

    // ---- phase 1: gather [xc | agg/cnt | u[batch]] ----
    {
        int r = tid >> 1, half = tid & 1;
        int n = base + r;
        unsigned short* row = sA + r * 264;
        if (n < n_nodes) {
            if (half == 0) {
                const unsigned short* px = xc + (size_t)n * 128;
#pragma unroll
                for (int j = 0; j < 128; j += 8)
                    *(u32x4*)(row + j) = *(const u32x4*)(px + j);
            } else {
                float c = cnt[n];
                float rs = 1.0f / (c > 1.0f ? c : 1.0f);
                const float* pa = agg + (size_t)n * 64;
#pragma unroll
                for (int j = 0; j < 64; j += 4) {
                    f32x4 a = *(const f32x4*)(pa + j);
                    row[128 + j + 0] = f2bf(a.x * rs); row[128 + j + 1] = f2bf(a.y * rs);
                    row[128 + j + 2] = f2bf(a.z * rs); row[128 + j + 3] = f2bf(a.w * rs);
                }
                int bn = batch[n];
                const unsigned short* pu = u_bf + (size_t)bn * 64;
#pragma unroll
                for (int j = 0; j < 64; j += 8)
                    *(u32x4*)(row + 192 + j) = *(const u32x4*)(pu + j);
                sB[r] = bn;
            }
        } else {
            u32x4 z = {0u, 0u, 0u, 0u};
            for (int j = half * 128; j < half * 128 + 128; j += 8) *(u32x4*)(row + j) = z;
            if (half) sB[r] = 0;
        }
    }
    __syncthreads();

    const int rowBase = wave * 16;

    // ---- node MLP GEMM1: 256 -> 128 ReLU ----
#pragma unroll
    for (int nt = 0; nt < 8; ++nt) {
        float bb = bn1[nt * 16 + (lane & 15)];
        v8f acc = {bb, bb, bb, bb, bb, bb, bb, bb};
#pragma unroll
        for (int kt = 0; kt < 8; ++kt) {
            v16bf a = load_a(sA, 264, rowBase, kt * 32, lane);
            v16bf b = load_b(Wn1p, kt * 8 + nt, lane);
            acc = wmma_bf16(a, b, acc);
        }
        int col = nt * 16 + (lane & 15);
        int r0 = rowBase + ((lane >> 4) << 3);
#pragma unroll
        for (int v = 0; v < 8; ++v) {
            float x = acc[v] > 0.f ? acc[v] : 0.f;
            sH[(r0 + v) * 136 + col] = f2bf(x);
        }
    }
    __syncthreads();

    // ---- node MLP GEMM2: 128 -> 64 ; keep result, restage for attention ----
    v8f xacc[4];
#pragma unroll
    for (int nt = 0; nt < 4; ++nt) {
        float bb = bn2[nt * 16 + (lane & 15)];
        v8f acc = {bb, bb, bb, bb, bb, bb, bb, bb};
#pragma unroll
        for (int kt = 0; kt < 4; ++kt) {
            v16bf a = load_a(sH, 136, rowBase, kt * 32, lane);
            v16bf b = load_b(Wn2p, kt * 4 + nt, lane);
            acc = wmma_bf16(a, b, acc);
        }
        xacc[nt] = acc;
        int col = nt * 16 + (lane & 15);
        int r0 = rowBase + ((lane >> 4) << 3);
#pragma unroll
        for (int v = 0; v < 8; ++v) {
            int m = r0 + v;
            int n = base + m;
            unsigned short bv = f2bf(acc[v]);
            if (n < n_nodes) x_new[(size_t)n * 64 + col] = bv;
            // overwrite agg columns so [x_new | u[batch]] is contiguous at col 128..255
            sA[m * 264 + 128 + col] = bv;
        }
    }
    __syncthreads();

    // ---- attention GEMM1: [x_new|u] 128 -> 128 ReLU ----
#pragma unroll
    for (int nt = 0; nt < 8; ++nt) {
        float bb = ba1[nt * 16 + (lane & 15)];
        v8f acc = {bb, bb, bb, bb, bb, bb, bb, bb};
#pragma unroll
        for (int kt = 0; kt < 4; ++kt) {
            v16bf a = load_a(sA, 264, rowBase, 128 + kt * 32, lane);
            v16bf b = load_b(Wa1p, kt * 8 + nt, lane);
            acc = wmma_bf16(a, b, acc);
        }
        int col = nt * 16 + (lane & 15);
        int r0 = rowBase + ((lane >> 4) << 3);
#pragma unroll
        for (int v = 0; v < 8; ++v) {
            float x = acc[v] > 0.f ? acc[v] : 0.f;
            sH[(r0 + v) * 136 + col] = f2bf(x);
        }
    }
    __syncthreads();

    // ---- attention GEMM2: 128 -> 64 ; sigmoid gate; scatter a*x_new into xg ----
#pragma unroll
    for (int nt = 0; nt < 4; ++nt) {
        float bb = ba2[nt * 16 + (lane & 15)];
        v8f acc = {bb, bb, bb, bb, bb, bb, bb, bb};
#pragma unroll
        for (int kt = 0; kt < 4; ++kt) {
            v16bf a = load_a(sH, 136, rowBase, kt * 32, lane);
            v16bf b = load_b(Wa2p, kt * 4 + nt, lane);
            acc = wmma_bf16(a, b, acc);
        }
        int col = nt * 16 + (lane & 15);
        int r0 = rowBase + ((lane >> 4) << 3);
#pragma unroll
        for (int v = 0; v < 8; ++v) {
            int m = r0 + v;
            int n = base + m;
            if (n < n_nodes) {
                float gate = 1.0f / (1.0f + __expf(-acc[v]));
                atomicAdd(xg + (size_t)sB[m] * 64 + col, gate * xacc[nt][v]);
            }
        }
    }
}

// ---------------- small 128-row MLP: out = W2(relu(W1 [srcA|srcB] + b1)) + b2 -------
__global__ __launch_bounds__(256) void mlp128_kernel(
    const float* __restrict__ srcA, const float* __restrict__ srcB,
    const unsigned short* __restrict__ W1p, const float* __restrict__ b1,
    const unsigned short* __restrict__ W2p, const float* __restrict__ b2,
    int FO, float* __restrict__ out) {

    __shared__ unsigned short sA[128 * 136];
    __shared__ unsigned short sH[128 * 136];

    const int tid = threadIdx.x;
    const int lane = tid & 31, wave = tid >> 5;

    {   // phase 1: build [srcA | srcB] bf16
        int r = tid >> 1, half = tid & 1;
        const float* s = half ? (srcB + (size_t)r * 64) : (srcA + (size_t)r * 64);
        unsigned short* row = sA + r * 136 + half * 64;
#pragma unroll
        for (int j = 0; j < 64; j += 4) {
            f32x4 a = *(const f32x4*)(s + j);
            row[j + 0] = f2bf(a.x); row[j + 1] = f2bf(a.y);
            row[j + 2] = f2bf(a.z); row[j + 3] = f2bf(a.w);
        }
    }
    __syncthreads();

    const int rowBase = wave * 16;
#pragma unroll
    for (int nt = 0; nt < 8; ++nt) {
        float bb = b1[nt * 16 + (lane & 15)];
        v8f acc = {bb, bb, bb, bb, bb, bb, bb, bb};
#pragma unroll
        for (int kt = 0; kt < 4; ++kt) {
            v16bf a = load_a(sA, 136, rowBase, kt * 32, lane);
            v16bf b = load_b(W1p, kt * 8 + nt, lane);
            acc = wmma_bf16(a, b, acc);
        }
        int col = nt * 16 + (lane & 15);
        int r0 = rowBase + ((lane >> 4) << 3);
#pragma unroll
        for (int v = 0; v < 8; ++v) {
            float x = acc[v] > 0.f ? acc[v] : 0.f;
            sH[(r0 + v) * 136 + col] = f2bf(x);
        }
    }
    __syncthreads();

    int ntiles2 = FO >> 4;
    for (int nt = 0; nt < ntiles2; ++nt) {
        float bb = b2[nt * 16 + (lane & 15)];
        v8f acc = {bb, bb, bb, bb, bb, bb, bb, bb};
#pragma unroll
        for (int kt = 0; kt < 4; ++kt) {
            v16bf a = load_a(sH, 136, rowBase, kt * 32, lane);
            v16bf b = load_b(W2p, kt * ntiles2 + nt, lane);
            acc = wmma_bf16(a, b, acc);
        }
        int col = nt * 16 + (lane & 15);
        int r0 = rowBase + ((lane >> 4) << 3);
#pragma unroll
        for (int v = 0; v < 8; ++v)
            out[(size_t)(r0 + v) * FO + col] = acc[v];
    }
}

// =====================================================================================
extern "C" void kernel_launch(void* const* d_in, const int* in_sizes, int n_in,
                              void* d_out, int out_size, void* d_ws, size_t ws_size,
                              hipStream_t stream) {
    // ---- inputs ----
    const float* x1  = (const float*)d_in[0];
    const float* e1  = (const float*)d_in[1];
    const float* u1  = (const float*)d_in[2];
    const int*   ei1 = (const int*)  d_in[3];
    const int*   bt1 = (const int*)  d_in[4];
    const float* x2  = (const float*)d_in[5];
    const float* e2  = (const float*)d_in[6];
    const float* u2  = (const float*)d_in[7];
    const int*   ei2 = (const int*)  d_in[8];
    const int*   bt2 = (const int*)  d_in[9];
    const float* We1 = (const float*)d_in[10]; const float* be1 = (const float*)d_in[11];
    const float* We2 = (const float*)d_in[12]; const float* be2 = (const float*)d_in[13];
    const float* Wn1 = (const float*)d_in[14]; const float* bn1 = (const float*)d_in[15];
    const float* Wn2 = (const float*)d_in[16]; const float* bn2 = (const float*)d_in[17];
    const float* Wa1 = (const float*)d_in[18]; const float* ba1 = (const float*)d_in[19];
    const float* Wa2 = (const float*)d_in[20]; const float* ba2 = (const float*)d_in[21];
    const float* Wg1 = (const float*)d_in[22]; const float* bg1 = (const float*)d_in[23];
    const float* Wg2 = (const float*)d_in[24]; const float* bg2 = (const float*)d_in[25];
    const float* Wm1 = (const float*)d_in[26]; const float* bm1 = (const float*)d_in[27];
    const float* Wm2 = (const float*)d_in[28]; const float* bm2 = (const float*)d_in[29];
    float* out = (float*)d_out;

    // ---- workspace carve (~270 MB) ----
    char* p = (char*)d_ws;
    auto alloc = [&](size_t bytes) -> void* {
        void* r = (void*)p; p += (bytes + 255) & ~(size_t)255; return r;
    };
    unsigned short* xc1 = (unsigned short*)alloc((size_t)NN * 128 * 2);
    unsigned short* xc2 = (unsigned short*)alloc((size_t)NN * 128 * 2);
    unsigned short* e1b = (unsigned short*)alloc((size_t)NE * 64 * 2);
    unsigned short* e2b = (unsigned short*)alloc((size_t)NE * 64 * 2);
    float* agg1 = (float*)alloc((size_t)NN * 64 * 4);
    float* agg2 = (float*)alloc((size_t)NN * 64 * 4);
    float* cnt1 = (float*)alloc((size_t)NN * 4);
    float* cnt2 = (float*)alloc((size_t)NN * 4);
    unsigned short* x1n = (unsigned short*)alloc((size_t)NN * 64 * 2);
    unsigned short* x2n = (unsigned short*)alloc((size_t)NN * 64 * 2);
    float* xg1  = (float*)alloc((size_t)NB * 64 * 4);
    float* xg2  = (float*)alloc((size_t)NB * 64 * 4);
    float* u1b  = (float*)alloc((size_t)NB * 64 * 4);
    float* u2b  = (float*)alloc((size_t)NB * 64 * 4);
    unsigned short* u1bf = (unsigned short*)alloc((size_t)NB * 64 * 2);
    unsigned short* u2bf = (unsigned short*)alloc((size_t)NB * 64 * 2);
    unsigned short* We1p = (unsigned short*)alloc(256 * 128 * 2);
    unsigned short* We2p = (unsigned short*)alloc(128 * 64 * 2);
    unsigned short* Wn1p = (unsigned short*)alloc(256 * 128 * 2);
    unsigned short* Wn2p = (unsigned short*)alloc(128 * 64 * 2);
    unsigned short* Wa1p = (unsigned short*)alloc(128 * 128 * 2);
    unsigned short* Wa2p = (unsigned short*)alloc(128 * 64 * 2);
    unsigned short* Wg1p = (unsigned short*)alloc(128 * 128 * 2);
    unsigned short* Wg2p = (unsigned short*)alloc(128 * 64 * 2);
    unsigned short* Wm1p = (unsigned short*)alloc(128 * 128 * 2);
    unsigned short* Wm2p = (unsigned short*)alloc(128 * 32 * 2);

    // ---- pack weights to bf16 WMMA-B fragment layout (weights stay hot in L2) ----
    pack_w_kernel<<<8 * 8, 32, 0, stream>>>(We1, We1p, 256, 128);
    pack_w_kernel<<<4 * 4, 32, 0, stream>>>(We2, We2p, 128, 64);
    pack_w_kernel<<<8 * 8, 32, 0, stream>>>(Wn1, Wn1p, 256, 128);
    pack_w_kernel<<<4 * 4, 32, 0, stream>>>(Wn2, Wn2p, 128, 64);
    pack_w_kernel<<<4 * 8, 32, 0, stream>>>(Wa1, Wa1p, 128, 128);
    pack_w_kernel<<<4 * 4, 32, 0, stream>>>(Wa2, Wa2p, 128, 64);
    pack_w_kernel<<<4 * 8, 32, 0, stream>>>(Wg1, Wg1p, 128, 128);
    pack_w_kernel<<<4 * 4, 32, 0, stream>>>(Wg2, Wg2p, 128, 64);
    pack_w_kernel<<<4 * 8, 32, 0, stream>>>(Wm1, Wm1p, 128, 128);
    pack_w_kernel<<<4 * 2, 32, 0, stream>>>(Wm2, Wm2p, 128, 32);

    // ---- init u buffers (f32 working copy + bf16 gather mirror) ----
    const int gU = (NB * 64 + 255) / 256;
    copy_f32_kernel<<<gU, 256, 0, stream>>>(u1, u1b, NB * 64);
    copy_f32_kernel<<<gU, 256, 0, stream>>>(u2, u2b, NB * 64);
    cvt_f32_bf16_kernel<<<gU, 256, 0, stream>>>(u1b, u1bf, NB * 64);
    cvt_f32_bf16_kernel<<<gU, 256, 0, stream>>>(u2b, u2bf, NB * 64);

    const int gXC  = (NN * 128 + 255) / 256;
    const int gAGG = (NN * 64 + 255) / 256;
    const int gCNT = (NN + 255) / 256;
    const int gXG  = (NB * 64 + 255) / 256;
    const int gE   = (NE + 63) / 64;
    const int gN   = (NN + 63) / 64;

    for (int s = 0; s < 2; ++s) {
        // ================= graph 1 (conditioned on current u2) =================
        build_xc_kernel<<<gXC, 256, 0, stream>>>(x1, x1n, s > 0, u2b, bt1, xc1, NN);
        zero_f32_kernel<<<gAGG, 256, 0, stream>>>(agg1, NN * 64);
        zero_f32_kernel<<<gCNT, 256, 0, stream>>>(cnt1, NN);
        zero_f32_kernel<<<gXG, 256, 0, stream>>>(xg1, NB * 64);
        edge_kernel<<<gE, 128, 0, stream>>>(xc1, e1, e1b, s > 0, u1bf, ei1, bt1,
                                            We1p, be1, We2p, be2, e1b, agg1, cnt1, NE);
        node_kernel<<<gN, 128, 0, stream>>>(xc1, agg1, cnt1, u1bf, bt1,
                                            Wn1p, bn1, Wn2p, bn2, Wa1p, ba1, Wa2p, ba2,
                                            x1n, xg1, NN);
        mlp128_kernel<<<1, 256, 0, stream>>>(xg1, u1b, Wg1p, bg1, Wg2p, bg2, 64, u1b);
        cvt_f32_bf16_kernel<<<gU, 256, 0, stream>>>(u1b, u1bf, NB * 64);

        // ================= graph 2 (conditioned on updated u1) =================
        build_xc_kernel<<<gXC, 256, 0, stream>>>(x2, x2n, s > 0, u1b, bt2, xc2, NN);
        zero_f32_kernel<<<gAGG, 256, 0, stream>>>(agg2, NN * 64);
        zero_f32_kernel<<<gCNT, 256, 0, stream>>>(cnt2, NN);
        zero_f32_kernel<<<gXG, 256, 0, stream>>>(xg2, NB * 64);
        edge_kernel<<<gE, 128, 0, stream>>>(xc2, e2, e2b, s > 0, u2bf, ei2, bt2,
                                            We1p, be1, We2p, be2, e2b, agg2, cnt2, NE);
        node_kernel<<<gN, 128, 0, stream>>>(xc2, agg2, cnt2, u2bf, bt2,
                                            Wn1p, bn1, Wn2p, bn2, Wa1p, ba1, Wa2p, ba2,
                                            x2n, xg2, NN);
        mlp128_kernel<<<1, 256, 0, stream>>>(xg2, u2b, Wg1p, bg1, Wg2p, bg2, 64, u2b);
        cvt_f32_bf16_kernel<<<gU, 256, 0, stream>>>(u2b, u2bf, NB * 64);

        // ================= readout =================
        mlp128_kernel<<<1, 256, 0, stream>>>(u1b, u2b, Wm1p, bm1, Wm2p, bm2, 32,
                                             out + (size_t)s * NB * 32);
    }
    (void)in_sizes; (void)n_in; (void)out_size; (void)ws_size;
}
